// VWVGraph_25933012533387
// MI455X (gfx1250) — compile-verified, matrix operand order checked by reading, not run on hardware
//
#include <hip/hip_runtime.h>
#include <hip/hip_bf16.h>
#include <math.h>

typedef _Float16 v16h __attribute__((ext_vector_type(16)));
typedef _Float16 v8h  __attribute__((ext_vector_type(8)));
typedef float    v8f  __attribute__((ext_vector_type(8)));
typedef float    v4f  __attribute__((ext_vector_type(4)));

#define N_ 1024
#define D_ 128

// ---------------------------------------------------------------------------
// Prep: per row i -> inv-norm f16 copy of V (for WMMA), a = V@W1[:D] + b1,
// b = V@W1[D:]. One block (128 threads) per row.
// ---------------------------------------------------------------------------
__global__ __launch_bounds__(128)
void vwv_prep_kernel(const float* __restrict__ V,
                     const float* __restrict__ W1,
                     const float* __restrict__ b1,
                     _Float16* __restrict__ Vnh,
                     float* __restrict__ a_out,
                     float* __restrict__ b_out) {
    __shared__ float vsh[D_];
    __shared__ float red[D_];
    const int i = blockIdx.x;
    const int t = threadIdx.x;
    const float v = V[i * D_ + t];
    vsh[t] = v;
    red[t] = v * v;
    __syncthreads();
    #pragma unroll
    for (int s = 64; s > 0; s >>= 1) {
        if (t < s) red[t] += red[t + s];
        __syncthreads();
    }
    const float nrm = fmaxf(sqrtf(red[0]), 1e-12f);   // F.normalize eps
    Vnh[i * D_ + t] = (_Float16)(v * (1.0f / nrm));

    float accA = 0.f, accB = 0.f;
    #pragma unroll 4
    for (int d = 0; d < D_; ++d) {
        const float vd = vsh[d];
        accA = fmaf(vd, W1[d * D_ + t], accA);          // W1[:D]
        accB = fmaf(vd, W1[(d + D_) * D_ + t], accB);   // W1[D:]
    }
    a_out[i * D_ + t] = accA + b1[t];                   // fold b1 into a
    b_out[i * D_ + t] = accB;
}

// ---------------------------------------------------------------------------
// softmax(alpha) head-fusion scalar (sums to ~1.0, computed for fidelity)
// ---------------------------------------------------------------------------
__global__ void vwv_scale_kernel(const float* __restrict__ alpha,
                                 float* __restrict__ scale) {
    if (threadIdx.x == 0) {
        float m = alpha[0];
        for (int k = 1; k < 4; ++k) m = fmaxf(m, alpha[k]);
        float e[4], s = 0.f;
        for (int k = 0; k < 4; ++k) { e[k] = __expf(alpha[k] - m); s += e[k]; }
        float tot = 0.f;
        for (int k = 0; k < 4; ++k) tot += e[k] / s;
        scale[0] = tot;
    }
}

// ---------------------------------------------------------------------------
// Fused tile kernel: one wave32 per 16x16 output tile.
//   Gram: v_wmma_f32_16x16x32_f16 chained over D=128 (4 WMMAs)
//   Mask: per-lane fused ReLU-MLP over LDS-staged a/b tiles
// ---------------------------------------------------------------------------
__global__ __launch_bounds__(32)
void vwv_fused_kernel(const _Float16* __restrict__ Vnh,
                      const float* __restrict__ a_g,
                      const float* __restrict__ b_g,
                      const float* __restrict__ W2,
                      const float* __restrict__ b2,
                      const float* __restrict__ scale_p,
                      float* __restrict__ out) {
    __shared__ float a_s[16 * 132];   // stride 132: 16B-aligned rows, conflict-free b128 reads
    __shared__ float b_s[16 * 132];
    __shared__ float w2_s[D_];

    const int lane = threadIdx.x;
    const int i0 = blockIdx.y << 4;
    const int j0 = blockIdx.x << 4;

    // ---- stage a/b tiles + W2 into LDS (vectorized, no divergence) ----
    #pragma unroll
    for (int idx = 0; idx < 16; ++idx) {
        const int flat = idx * 32 + lane;       // 512 float4 slots
        const int row = flat >> 5;
        const int c4  = (flat & 31) << 2;
        *(v4f*)&a_s[row * 132 + c4] = *(const v4f*)&a_g[(i0 + row) * D_ + c4];
        *(v4f*)&b_s[row * 132 + c4] = *(const v4f*)&b_g[(j0 + row) * D_ + c4];
    }
    *(v4f*)&w2_s[lane << 2] = *(const v4f*)&W2[lane << 2];
    __syncthreads();

    // ---- Gram tile via WMMA: A lane=M holds K regs; B = Vn^T has the SAME
    //      per-lane layout (lane=N holds K regs), so both load straight from
    //      row-major f16 Vn with two b128 loads per K-block. ----
    const int r16 = lane & 15;            // M-lane for A / N-lane for B / N of C
    const int kb8 = (lane >> 4) << 3;     // upper half-wave owns K+8 chunk
    const _Float16* arow = Vnh + (i0 + r16) * D_;
    const _Float16* brow = Vnh + (j0 + r16) * D_;
    v8f c = {};
    #pragma unroll
    for (int kb = 0; kb < 4; ++kb) {
        const int k0 = kb * 32 + kb8;
        union { v16h v; v8h h[2]; } ua, ub;
        ua.h[0] = *(const v8h*)(arow + k0);
        ua.h[1] = *(const v8h*)(arow + k0 + 16);
        ub.h[0] = *(const v8h*)(brow + k0);
        ub.h[1] = *(const v8h*)(brow + k0 + 16);
        c = __builtin_amdgcn_wmma_f32_16x16x32_f16(
                /*neg_a=*/false, ua.v, /*neg_b=*/false, ub.v,
                /*c_mod=*/(short)0, c, /*reuse_a=*/false, /*reuse_b=*/false);
    }

    // ---- fused edge-MLP: acc[r] = sum_d relu(a[m][d]+b[n][d]) * W2[d]
    //      C layout: lanes 0-15 -> M=r, lanes 16-31 -> M=r+8; N = lane%16 ----
    const int mo = (lane >> 4) << 3;
    float acc[8] = {0.f, 0.f, 0.f, 0.f, 0.f, 0.f, 0.f, 0.f};
    for (int d = 0; d < D_; d += 4) {
        const v4f w2 = *(const v4f*)&w2_s[d];
        const v4f bb = *(const v4f*)&b_s[r16 * 132 + d];
        #pragma unroll
        for (int r = 0; r < 8; ++r) {
            const v4f aa = *(const v4f*)&a_s[(mo + r) * 132 + d];
            acc[r] = fmaf(fmaxf(aa.x + bb.x, 0.f), w2.x, acc[r]);
            acc[r] = fmaf(fmaxf(aa.y + bb.y, 0.f), w2.y, acc[r]);
            acc[r] = fmaf(fmaxf(aa.z + bb.z, 0.f), w2.z, acc[r]);
            acc[r] = fmaf(fmaxf(aa.w + bb.w, 0.f), w2.w, acc[r]);
        }
    }

    // ---- fuse: out = (cos_sim^2) * sigmoid(mlp + b2) * softmax_sum ----
    // sigmoid via hardware v_rcp_f32: exact IEEE divide is pointless after
    // the v_exp_f32 approximation and its div_scale/div_fixup chain would
    // serialize the epilogue.
    const float b2v = b2[0];
    const float sc  = scale_p[0];
    #pragma unroll
    for (int r = 0; r < 8; ++r) {
        const float s    = c[r];
        const float mask = __builtin_amdgcn_rcpf(1.f + __expf(-(acc[r] + b2v)));
        out[(i0 + mo + r) * N_ + (j0 + r16)] = s * s * mask * sc;
    }
}

// ---------------------------------------------------------------------------
// inputs: V(131072) Wk(65536, unused) alpha(4) W1(32768) b1(128) W2(128) b2(1)
// ---------------------------------------------------------------------------
extern "C" void kernel_launch(void* const* d_in, const int* in_sizes, int n_in,
                              void* d_out, int out_size, void* d_ws, size_t ws_size,
                              hipStream_t stream) {
    const float* V     = (const float*)d_in[0];
    const float* alpha = (const float*)d_in[2];
    const float* W1    = (const float*)d_in[3];
    const float* b1    = (const float*)d_in[4];
    const float* W2    = (const float*)d_in[5];
    const float* b2    = (const float*)d_in[6];
    float* out = (float*)d_out;

    char* ws = (char*)d_ws;
    _Float16* Vnh = (_Float16*)ws;                                 // 256 KB
    float* a_g     = (float*)(ws + (size_t)N_ * D_ * 2);           // 512 KB
    float* b_g     = (float*)(ws + (size_t)N_ * D_ * 2 + (size_t)N_ * D_ * 4);
    float* scale_p = (float*)(ws + (size_t)N_ * D_ * 2 + (size_t)N_ * D_ * 8);

    vwv_prep_kernel<<<dim3(N_), dim3(D_), 0, stream>>>(V, W1, b1, Vnh, a_g, b_g);
    vwv_scale_kernel<<<dim3(1), dim3(32), 0, stream>>>(alpha, scale_p);
    vwv_fused_kernel<<<dim3(N_ / 16, N_ / 16), dim3(32), 0, stream>>>(
        Vnh, a_g, b_g, W2, b2, scale_p, out);
}
